// GMNLayer_87445534147346
// MI455X (gfx1250) — matrix operand order repaired and checked by reading, not compiled, not run on hardware
//
#include <hip/hip_runtime.h>

typedef __attribute__((ext_vector_type(16))) _Float16 v16h;
typedef __attribute__((ext_vector_type(8)))  _Float16 v8h;
typedef __attribute__((ext_vector_type(4)))  _Float16 v4h;
typedef __attribute__((ext_vector_type(8)))  float    v8f;

__device__ __forceinline__ float silu_f(float x) { return x / (1.f + __expf(-x)); }

__device__ __forceinline__ v8f zero8() {
    v8f r;
    #pragma unroll
    for (int i = 0; i < 8; ++i) r[i] = 0.f;
    return r;
}

// Build a 16x32 f16 A-fragment from a row-major LDS buffer (stride in halves).
// CDNA5 ISA 7.12.2: lanes 0-15 (M=lane) hold K = k0+0..7 / k0+16..23,
// lanes 16-31 (M=lane&15) hold K = k0+8..15 / k0+24..31.
__device__ __forceinline__ v16h a_frag(const _Float16* base, int astride, int m0, int k0) {
    const int lane = threadIdx.x & 31;
    const _Float16* p = base + (m0 + (lane & 15)) * astride + k0 + ((lane & 16) >> 1);
    union { v16h v; v8h h[2]; } u;
    u.h[0] = *(const v8h*)(p);
    u.h[1] = *(const v8h*)(p + 16);
    return u.v;
}

// One wave computes a 64-row x 16-col output strip: 4 M-tiles reusing the
// pre-swizzled B fragment (frag-contiguous, 512 halves each).
template <int KT, int NT>
__device__ __forceinline__ void gemm_col4(const _Float16* A, int astride,
                                          const _Float16* Bw, int ntile, v8f acc[4]) {
    const int lane = threadIdx.x & 31;
    #pragma unroll
    for (int kt = 0; kt < KT; ++kt) {
        v16h b = *(const v16h*)(Bw + ((kt * NT + ntile) << 9) + (lane << 4));
        #pragma unroll
        for (int mt = 0; mt < 4; ++mt) {
            v16h a = a_frag(A, astride, mt * 16, kt * 32);
            acc[mt] = __builtin_amdgcn_wmma_f32_16x16x32_f16(
                false, a, false, b, (short)0, acc[mt], false, false);
        }
    }
}

template <int KT, int NT>
__device__ __forceinline__ v8f gemm_tile(const _Float16* A, int astride,
                                         const _Float16* Bw, int mtile, int ntile) {
    const int lane = threadIdx.x & 31;
    v8f acc = zero8();
    #pragma unroll
    for (int kt = 0; kt < KT; ++kt) {
        v16h b = *(const v16h*)(Bw + ((kt * NT + ntile) << 9) + (lane << 4));
        v16h a = a_frag(A, astride, mtile * 16, kt * 32);
        acc = __builtin_amdgcn_wmma_f32_16x16x32_f16(
            false, a, false, b, (short)0, acc, false, false);
    }
    return acc;
}

// f32 weight [K x Nn] row-major -> f16 WMMA B-fragments:
// out[frag*512 + lane*16 + j], frag = kt*NT + nt,
// value = W[kt*32 + (j&7) + (lane>=16?8:0) + (j>=8?16:0)][nt*16 + (lane&15)], zero padded.
__global__ void gmn_swizzle_w(const float* __restrict__ W, _Float16* __restrict__ out,
                              int K, int Nn, int KT, int NT) {
    const int total = KT * NT * 512;
    for (int idx = blockIdx.x * blockDim.x + threadIdx.x; idx < total;
         idx += gridDim.x * blockDim.x) {
        const int j    = idx & 15;
        const int lane = (idx >> 4) & 31;
        const int frag = idx >> 9;
        const int nt = frag % NT;
        const int kt = frag / NT;
        const int k = kt * 32 + (j & 7) + ((lane >> 4) << 3) + ((j >> 3) << 4);
        const int n = nt * 16 + (lane & 15);
        float v = (k < K && n < Nn) ? W[k * Nn + n] : 0.f;
        out[idx] = (_Float16)v;
    }
}

__global__ void gmn_zero_f32(float* __restrict__ p, long n) {
    long i = (long)blockIdx.x * blockDim.x + threadIdx.x;
    const long stride = (long)gridDim.x * blockDim.x;
    for (; i < n; i += stride) p[i] = 0.f;
}

// LDS layout (edge kernel), bytes:
//   xh     @ 0      : 64 x 512 f16  (65536)   [overlaid by basisf after GEMM1]
//   bufA   @ 65536  : 64 x 128 f16  (16384)
//   bufB   @ 81920  : 64 x 128 f16  (16384)
//   Zf     @ 98304  : 64 x 60  f32  (15360)
//   invtmp @ 113664 : 64 x 100 f32  (25600)
//   sstmp  @ 139264 : 64 x 4   f32  (1024)
//   dsts   @ 140288 : 64 int        (256)
//   edfh   @ 140544 : 64 x 32  f16  (4096)
#define EDGE_LDS_BYTES 144640

__global__ __launch_bounds__(256) void gmn_edge_kernel(
    const float* __restrict__ Z, const float* __restrict__ h,
    const int* __restrict__ ei, const float* __restrict__ edf,
    const float* __restrict__ evec,
    const float* __restrict__ be,
    const float* __restrict__ bm1, const float* __restrict__ bm2,
    const float* __restrict__ bv1, const float* __restrict__ bv2,
    const _Float16* __restrict__ Weh,
    const _Float16* __restrict__ Wm1h, const _Float16* __restrict__ Wm2h,
    const _Float16* __restrict__ Wv1h, const _Float16* __restrict__ Wv2h,
    float* __restrict__ Zacc, float* __restrict__ macc, float* __restrict__ cnt,
    int Etot) {
    extern __shared__ char smem[];
    _Float16* xh     = (_Float16*)smem;
    _Float16* bufA   = (_Float16*)(smem + 65536);
    _Float16* bufB   = (_Float16*)(smem + 81920);
    float*    Zf     = (float*)(smem + 98304);
    float*    invtmp = (float*)(smem + 113664);
    float*    sstmp  = (float*)(smem + 139264);
    int*      dsts   = (int*)(smem + 140288);
    _Float16* edfh   = (_Float16*)(smem + 140544);
    float*    basisf = (float*)smem;   // 64 x 80 f32, overlays xh after GEMM1

    const int tid  = threadIdx.x;
    const int lane = tid & 31;
    const int wave = tid >> 5;
    const int e0   = blockIdx.x * 64;

    const int e  = tid >> 2;
    const int t4 = tid & 3;
    const int eg = e0 + e;
    const bool valid = eg < Etot;
    const int srcn = valid ? ei[eg] : 0;
    const int dstn = valid ? ei[Etot + eg] : 0;
    if (t4 == 0) dsts[e] = valid ? dstn : -1;

    // ---- Phase A1: gathers + edf staging + Z_ij fill (vectorized) ----
    for (int c8 = t4 * 32; c8 < t4 * 32 + 32; c8 += 8) {
        union { v8h v; _Float16 s[8]; } ui, uj;
        #pragma unroll
        for (int q = 0; q < 8; ++q) { ui.s[q] = (_Float16)0.f; uj.s[q] = (_Float16)0.f; }
        if (valid) {
            const float4 a0 = *(const float4*)(h + (size_t)dstn * 128 + c8);
            const float4 a1 = *(const float4*)(h + (size_t)dstn * 128 + c8 + 4);
            const float4 b0 = *(const float4*)(h + (size_t)srcn * 128 + c8);
            const float4 b1 = *(const float4*)(h + (size_t)srcn * 128 + c8 + 4);
            ui.s[0] = (_Float16)a0.x; ui.s[1] = (_Float16)a0.y;
            ui.s[2] = (_Float16)a0.z; ui.s[3] = (_Float16)a0.w;
            ui.s[4] = (_Float16)a1.x; ui.s[5] = (_Float16)a1.y;
            ui.s[6] = (_Float16)a1.z; ui.s[7] = (_Float16)a1.w;
            uj.s[0] = (_Float16)b0.x; uj.s[1] = (_Float16)b0.y;
            uj.s[2] = (_Float16)b0.z; uj.s[3] = (_Float16)b0.w;
            uj.s[4] = (_Float16)b1.x; uj.s[5] = (_Float16)b1.y;
            uj.s[6] = (_Float16)b1.z; uj.s[7] = (_Float16)b1.w;
        }
        *(v8h*)(xh + e * 512 + c8)       = ui.v;        // h_i (dst)
        *(v8h*)(xh + e * 512 + 128 + c8) = uj.v;        // h_j (src)
    }
    { // edf staging into LDS [64 x 32] f16 (K padded 16->32 with zeros)
        union { v8h v; _Float16 s[8]; } u;
        #pragma unroll
        for (int q = 0; q < 8; ++q) u.s[q] = (_Float16)0.f;
        if (t4 < 2 && valid) {
            const float4 a0 = *(const float4*)(edf + (size_t)eg * 16 + t4 * 8);
            const float4 a1 = *(const float4*)(edf + (size_t)eg * 16 + t4 * 8 + 4);
            u.s[0] = (_Float16)a0.x; u.s[1] = (_Float16)a0.y;
            u.s[2] = (_Float16)a0.z; u.s[3] = (_Float16)a0.w;
            u.s[4] = (_Float16)a1.x; u.s[5] = (_Float16)a1.y;
            u.s[6] = (_Float16)a1.z; u.s[7] = (_Float16)a1.w;
        }
        *(v8h*)(edfh + e * 32 + t4 * 8) = u.v;
    }
    if (t4 == 3) { // zero-pad cols 484..511
        union { v4h v; _Float16 s[4]; } z;
        #pragma unroll
        for (int q = 0; q < 4; ++q) z.s[q] = (_Float16)0.f;
        #pragma unroll
        for (int c = 484; c < 512; c += 4) *(v4h*)(xh + e * 512 + c) = z.v;
    }
    if (t4 < 3) { // Z_ij[d=t4][5][4]: diffs for t<4, broadcast edge vec at t=4
        const int d = t4;
        float4 z0 = {0.f, 0.f, 0.f, 0.f}, z1 = z0, z2 = z0, z3 = z0;
        float ev = 0.f;
        if (valid) {
            const float* zd = Z + (size_t)dstn * 48 + d * 16;
            const float* zs = Z + (size_t)srcn * 48 + d * 16;
            #pragma unroll
            for (int q = 0; q < 4; ++q) {
                float4 a = *(const float4*)(zd + q * 4);
                float4 b = *(const float4*)(zs + q * 4);
                float4 r; r.x = a.x - b.x; r.y = a.y - b.y; r.z = a.z - b.z; r.w = a.w - b.w;
                if (q == 0) z0 = r; else if (q == 1) z1 = r; else if (q == 2) z2 = r; else z3 = r;
            }
            ev = evec[(size_t)eg * 3 + d];
        }
        float* zrow = Zf + e * 60 + d * 20;
        *(float4*)(zrow + 0)  = z0;
        *(float4*)(zrow + 4)  = z1;
        *(float4*)(zrow + 8)  = z2;
        *(float4*)(zrow + 12) = z3;
        float4 evv; evv.x = ev; evv.y = ev; evv.z = ev; evv.w = ev;
        *(float4*)(zrow + 16) = evv;
    }
    __syncthreads();

    // ---- Phase A2: Gram invariants, parallel over 4 threads/edge ----
    {
        float ssq = 0.f;
        for (int p = t4; p < 25; p += 4) {
            const int tt = p / 5, rr = p % 5;
            float s0 = 0.f, s1 = 0.f, s2 = 0.f, s3 = 0.f;
            #pragma unroll
            for (int d = 0; d < 3; ++d) {
                const float* za = Zf + e * 60 + d * 20 + tt * 4;
                const float* zb = Zf + e * 60 + d * 20 + rr * 4;
                s0 += za[0] * zb[0]; s1 += za[1] * zb[1];
                s2 += za[2] * zb[2]; s3 += za[3] * zb[3];
            }
            float4 sv; sv.x = s0; sv.y = s1; sv.z = s2; sv.w = s3;
            *(float4*)(invtmp + e * 100 + p * 4) = sv;
            ssq += s0 * s0 + s1 * s1 + s2 * s2 + s3 * s3;
        }
        sstmp[e * 4 + t4] = ssq;
    }
    __syncthreads();

    const int nloc = lane & 15;
    const int mhi  = (lane & 16) >> 1;   // +8 rows for lanes 16..31 (C layout)

    // ---- Phase A3: normalize invariants -> xh[256..355]; edge_fc via WMMA
    //      -> xh[356..483]. Disjoint column ranges between the same barriers.
    {
        const float tot = sstmp[e * 4 + 0] + sstmp[e * 4 + 1] +
                          sstmp[e * 4 + 2] + sstmp[e * 4 + 3];
        const float rinv = 1.f / fmaxf(sqrtf(tot), 1e-12f);
        for (int p = t4; p < 25; p += 4) {
            union { v4h v; _Float16 s[4]; } u;
            #pragma unroll
            for (int q = 0; q < 4; ++q)
                u.s[q] = (_Float16)(invtmp[e * 100 + p * 4 + q] * rinv);
            *(v4h*)(xh + e * 512 + 256 + p * 4) = u.v;
        }
    }
    { // edge_fc: ef = edf @ We + be  (K=16 padded to 32, one WMMA per M-tile)
        v8f acc[4]; for (int i = 0; i < 4; ++i) acc[i] = zero8();
        gemm_col4<1, 8>(edfh, 32, Weh, wave, acc);
        const int col = wave * 16 + nloc;
        const float bb = be[col];
        #pragma unroll
        for (int mt = 0; mt < 4; ++mt)
            #pragma unroll
            for (int vr = 0; vr < 8; ++vr) {
                const int row = mt * 16 + vr + mhi;
                xh[row * 512 + 356 + col] = (_Float16)(acc[mt][vr] + bb);
            }
    }
    __syncthreads();

    // ---- Phase B: t1 = silu(x @ Wm1 + bm1) ----
    {
        v8f acc[4]; for (int i = 0; i < 4; ++i) acc[i] = zero8();
        gemm_col4<16, 8>(xh, 512, Wm1h, wave, acc);
        const int col = wave * 16 + nloc;
        const float bb = bm1[col];
        #pragma unroll
        for (int mt = 0; mt < 4; ++mt)
            #pragma unroll
            for (int vr = 0; vr < 8; ++vr) {
                const int row = mt * 16 + vr + mhi;
                bufA[row * 128 + col] = (_Float16)silu_f(acc[mt][vr] + bb);
            }
    }
    __syncthreads();

    // ---- Phase C: msg = silu(t1 @ Wm2 + bm2); scatter-add msg ----
    {
        v8f acc[4]; for (int i = 0; i < 4; ++i) acc[i] = zero8();
        gemm_col4<4, 8>(bufA, 128, Wm2h, wave, acc);
        const int col = wave * 16 + nloc;
        const float bb = bm2[col];
        #pragma unroll
        for (int mt = 0; mt < 4; ++mt)
            #pragma unroll
            for (int vr = 0; vr < 8; ++vr) {
                const int row = mt * 16 + vr + mhi;
                const float s = silu_f(acc[mt][vr] + bb);
                bufB[row * 128 + col] = (_Float16)s;
                const int dn = dsts[row];
                if (dn >= 0) atomicAdd(&macc[(size_t)dn * 128 + col], s);
            }
    }
    __syncthreads();

    // ---- Phase D: v1 = silu(msg @ Wv1 + bv1) ----
    {
        v8f acc[4]; for (int i = 0; i < 4; ++i) acc[i] = zero8();
        gemm_col4<4, 8>(bufB, 128, Wv1h, wave, acc);
        const int col = wave * 16 + nloc;
        const float bb = bv1[col];
        #pragma unroll
        for (int mt = 0; mt < 4; ++mt)
            #pragma unroll
            for (int vr = 0; vr < 8; ++vr) {
                const int row = mt * 16 + vr + mhi;
                bufA[row * 128 + col] = (_Float16)silu_f(acc[mt][vr] + bb);
            }
    }
    __syncthreads();

    // ---- Phase E: basis = v1 @ Wv2 + bv2 (5 ntiles x 4 mtiles) ----
    for (int tidx = wave; tidx < 20; tidx += 8) {
        const int mtile = tidx & 3;
        const int ntile = tidx >> 2;
        v8f acc = gemm_tile<4, 5>(bufA, 128, Wv2h, mtile, ntile);
        const int col = ntile * 16 + nloc;
        const float bb = bv2[col];
        #pragma unroll
        for (int vr = 0; vr < 8; ++vr) {
            const int row = mtile * 16 + vr + mhi;
            basisf[row * 80 + col] = acc[vr] + bb;
        }
    }
    __syncthreads();

    // ---- Phase F: Z_agg[d,k,h] = sum_t Z_ij[d,t,h]*basis[t,k,h]; scatter ----
    {
        const int dn = dsts[e];
        if (dn >= 0) {
            #pragma unroll
            for (int oo = 0; oo < 12; ++oo) {
                const int o  = t4 * 12 + oo;
                const int d  = o >> 4;
                const int k  = (o >> 2) & 3;
                const int hh = o & 3;
                float s = 0.f;
                #pragma unroll
                for (int tt = 0; tt < 5; ++tt)
                    s += Zf[e * 60 + d * 20 + tt * 4 + hh] *
                         basisf[e * 80 + tt * 16 + k * 4 + hh];
                atomicAdd(&Zacc[(size_t)dn * 48 + o], s);
            }
            if (t4 == 0) atomicAdd(&cnt[dn], 1.f);
        }
    }
}

// ---------------- Node kernel: 64 nodes per 256-thread block ----------------
__global__ __launch_bounds__(256) void gmn_node_kernel(
    const float* __restrict__ h, const float* __restrict__ macc,
    const float* __restrict__ cnt,
    const float* __restrict__ bs1, const float* __restrict__ bs2,
    const _Float16* __restrict__ Ws1h, const _Float16* __restrict__ Ws2h,
    float* __restrict__ Zio, float* __restrict__ hout, int Ntot) {
    extern __shared__ char smem[];
    _Float16* xn = (_Float16*)smem;             // 64 x 256 f16 (32 KB)
    _Float16* th = (_Float16*)(smem + 32768);   // 64 x 128 f16 (16 KB)

    const int tid  = threadIdx.x;
    const int lane = tid & 31;
    const int wave = tid >> 5;
    const int n0   = blockIdx.x * 64;

    {
        const int nd = tid >> 2;
        const int t4 = tid & 3;
        const int ng = n0 + nd;
        const bool valid = ng < Ntot;
        // cols [t4*64, t4*64+64): t4<2 -> h, t4>=2 -> macc (aligned at 128 split)
        const float* srcp = (t4 < 2) ? (h + (size_t)ng * 128 + t4 * 64)
                                     : (macc + (size_t)ng * 128 + (t4 - 2) * 64);
        for (int c8 = 0; c8 < 64; c8 += 8) {
            union { v8h v; _Float16 s[8]; } u;
            #pragma unroll
            for (int q = 0; q < 8; ++q) u.s[q] = (_Float16)0.f;
            if (valid) {
                const float4 a0 = *(const float4*)(srcp + c8);
                const float4 a1 = *(const float4*)(srcp + c8 + 4);
                u.s[0] = (_Float16)a0.x; u.s[1] = (_Float16)a0.y;
                u.s[2] = (_Float16)a0.z; u.s[3] = (_Float16)a0.w;
                u.s[4] = (_Float16)a1.x; u.s[5] = (_Float16)a1.y;
                u.s[6] = (_Float16)a1.z; u.s[7] = (_Float16)a1.w;
            }
            *(v8h*)(xn + nd * 256 + t4 * 64 + c8) = u.v;
        }
        if (valid) {  // Z mean: divide accumulated sums by counts, in place
            const float r = 1.f / fmaxf(cnt[ng], 1.f);
            for (int o = t4 * 12; o < t4 * 12 + 12; ++o) Zio[(size_t)ng * 48 + o] *= r;
        }
    }
    __syncthreads();

    const int nloc = lane & 15;
    const int mhi  = (lane & 16) >> 1;

    { // t = silu([h|m] @ Ws1 + bs1)
        v8f acc[4]; for (int i = 0; i < 4; ++i) acc[i] = zero8();
        gemm_col4<8, 8>(xn, 256, Ws1h, wave, acc);
        const int col = wave * 16 + nloc;
        const float bb = bs1[col];
        #pragma unroll
        for (int mt = 0; mt < 4; ++mt)
            #pragma unroll
            for (int vr = 0; vr < 8; ++vr) {
                const int row = mt * 16 + vr + mhi;
                th[row * 128 + col] = (_Float16)silu_f(acc[mt][vr] + bb);
            }
    }
    __syncthreads();

    { // h_out = t @ Ws2 + bs2
        v8f acc[4]; for (int i = 0; i < 4; ++i) acc[i] = zero8();
        gemm_col4<4, 8>(th, 128, Ws2h, wave, acc);
        const int col = wave * 16 + nloc;
        const float bb = bs2[col];
        #pragma unroll
        for (int mt = 0; mt < 4; ++mt)
            #pragma unroll
            for (int vr = 0; vr < 8; ++vr) {
                const int row = mt * 16 + vr + mhi;
                const int ng = n0 + row;
                if (ng < Ntot) hout[(size_t)ng * 128 + col] = acc[mt][vr] + bb;
            }
    }
}

extern "C" void kernel_launch(void* const* d_in, const int* in_sizes, int n_in,
                              void* d_out, int out_size, void* d_ws, size_t ws_size,
                              hipStream_t stream) {
    (void)n_in; (void)out_size; (void)ws_size;
    const float* Z    = (const float*)d_in[0];
    const float* h    = (const float*)d_in[1];
    const int*   ei   = (const int*)d_in[2];
    const float* edf  = (const float*)d_in[3];
    const float* evec = (const float*)d_in[4];
    // d_in[5] = edge_distance (unused by reference)
    const float* We  = (const float*)d_in[6];
    const float* be  = (const float*)d_in[7];
    const float* Wm1 = (const float*)d_in[8];
    const float* bm1 = (const float*)d_in[9];
    const float* Wm2 = (const float*)d_in[10];
    const float* bm2 = (const float*)d_in[11];
    const float* Wv1 = (const float*)d_in[12];
    const float* bv1 = (const float*)d_in[13];
    const float* Wv2 = (const float*)d_in[14];
    const float* bv2 = (const float*)d_in[15];
    const float* Ws1 = (const float*)d_in[16];
    const float* bs1 = (const float*)d_in[17];
    const float* Ws2 = (const float*)d_in[18];
    const float* bs2 = (const float*)d_in[19];

    const int N = in_sizes[1] / 128;
    const int E = in_sizes[2] / 2;

    char* ws = (char*)d_ws;
    _Float16* Wm1h = (_Float16*)(ws + 0);        // 16*8*512 halves = 128 KB
    _Float16* Wm2h = (_Float16*)(ws + 131072);   //  32 KB
    _Float16* Wv1h = (_Float16*)(ws + 163840);   //  32 KB
    _Float16* Wv2h = (_Float16*)(ws + 196608);   //  20 KB
    _Float16* Ws1h = (_Float16*)(ws + 217088);   //  64 KB
    _Float16* Ws2h = (_Float16*)(ws + 282624);   //  32 KB
    _Float16* Weh  = (_Float16*)(ws + 315392);   //  1*8*512 halves = 8 KB
    float* macc = (float*)(ws + 323584);                          // N*128 f32
    float* cnt  = (float*)(ws + 323584 + (size_t)N * 128 * 4);    // N f32

    float* Zacc = (float*)d_out;                     // N*48 accumulated then meaned
    float* hout = (float*)d_out + (size_t)N * 48;    // N*128

    { // weight swizzles (cheap; rerun each call for determinism)
        struct SW { const float* W; _Float16* o; int K, Nn, KT, NT; } sws[7] = {
            {Wm1, Wm1h, 484, 128, 16, 8}, {Wm2, Wm2h, 128, 128, 4, 8},
            {Wv1, Wv1h, 128, 128, 4, 8},  {Wv2, Wv2h, 128,  80, 4, 5},
            {Ws1, Ws1h, 256, 128, 8, 8},  {Ws2, Ws2h, 128, 128, 4, 8},
            {We,  Weh,   16, 128, 1, 8}};
        for (int i = 0; i < 7; ++i) {
            int total = sws[i].KT * sws[i].NT * 512;
            int blocks = (total + 255) / 256;
            gmn_swizzle_w<<<blocks, 256, 0, stream>>>(sws[i].W, sws[i].o, sws[i].K,
                                                      sws[i].Nn, sws[i].KT, sws[i].NT);
        }
    }
    { // zero scatter accumulators
        struct ZR { float* p; long n; } zs[3] = {
            {Zacc, (long)N * 48}, {macc, (long)N * 128}, {cnt, (long)N}};
        for (int i = 0; i < 3; ++i) {
            int blocks = (int)((zs[i].n + 255) / 256);
            if (blocks > 4096) blocks = 4096;
            gmn_zero_f32<<<blocks, 256, 0, stream>>>(zs[i].p, zs[i].n);
        }
    }

    const int eblocks = (E + 63) / 64;
    gmn_edge_kernel<<<eblocks, 256, EDGE_LDS_BYTES, stream>>>(
        Z, h, ei, edf, evec, be, bm1, bm2, bv1, bv2,
        Weh, Wm1h, Wm2h, Wv1h, Wv2h, Zacc, macc, cnt, E);

    const int nblocks = (N + 63) / 64;
    gmn_node_kernel<<<nblocks, 256, 49152, stream>>>(
        h, macc, cnt, bs1, bs2, Ws1h, Ws2h, Zacc, hout, N);
}
